// ForcedGPULSTMCell_53137335386702
// MI455X (gfx1250) — matrix operand order repaired and checked by reading, not compiled, not run on hardware
//
#include <hip/hip_runtime.h>
#include <cstdint>
#include <cstddef>

// ---------------------------------------------------------------------------
// LSTM cell, MI455X (gfx1250).
//   gates = x @ Wx^T + h @ Wh^T + b  (two GEMMs, 275 GFLOP total)
//   bf16 WMMA (v_wmma_f32_16x16x32_bf16), f32 accumulate, fused gate epilogue.
// Pre-pass converts x, h and the 8 weight matrices to bf16 in d_ws (~100 MB,
// fits the 192 MB L2 so the GEMM K-loop streams from L2, not HBM).
//
// Wave mapping: each of the 4 waves owns one 16-wide column subtile of the
// 64x64 block, across all four 16-row M strips and all 4 gates. Per K-step a
// wave loads 4 A-fragments + 4 B-fragments (16 ds_load_b128) for 16 WMMAs —
// half the LDS fragment traffic of a per-M-strip mapping.
// ---------------------------------------------------------------------------

#define HID    2048
#define BSZ    4096
#define BM     64          // batch rows per workgroup
#define BN     64          // hidden cols per workgroup
#define BK     32          // K per WMMA step
#define LDSTR  40          // padded LDS row stride in bf16 elems (80B, 16B-aligned)
#define TILE_ELEMS  (64 * LDSTR)        // one 64x32 padded tile = 2560 elems
#define STAGE_ELEMS (5 * TILE_ELEMS)    // A tile + 4 gate B tiles = 12800 elems
#define NSTEPS 128                       // (2048 + 2048) / 32

typedef __attribute__((ext_vector_type(16))) __bf16 bf16x16;
typedef __attribute__((ext_vector_type(8)))  float  f32x8;

union Frag {
  bf16x16 v;
  uint4   q[2];
};

// Async per-lane 16B copy global -> LDS (ASYNCcnt-tracked, CDNA5 path).
__device__ __forceinline__ void async_cp16(uint32_t lds_off, const void* g) {
  asm volatile("global_load_async_to_lds_b128 %0, %1, off"
               :: "v"(lds_off), "v"((unsigned long long)(uintptr_t)g)
               : "memory");
}

__device__ __forceinline__ float fast_sigmoid(float x) {
  return 1.0f / (1.0f + __expf(-x));
}
__device__ __forceinline__ float fast_tanh(float x) {
  float e = __expf(2.0f * x);          // tanh(x) = 1 - 2/(e^{2x}+1)
  return 1.0f - 2.0f / (e + 1.0f);
}

// ---------------------------------------------------------------------------
// f32 -> bf16 conversion pre-pass (memory-bound, ~10us of HBM traffic total)
// ---------------------------------------------------------------------------
__global__ void __launch_bounds__(256)
cvt_f32_bf16(const float* __restrict__ s, __bf16* __restrict__ d, int n) {
  long long stride = (long long)gridDim.x * blockDim.x;
  for (long long i = (long long)blockIdx.x * blockDim.x + threadIdx.x;
       i * 4 < n; i += stride) {
    float4 v;
    __builtin_memcpy(&v, s + i * 4, 16);
    unsigned short u0 = __builtin_bit_cast(unsigned short, (__bf16)v.x);
    unsigned short u1 = __builtin_bit_cast(unsigned short, (__bf16)v.y);
    unsigned short u2 = __builtin_bit_cast(unsigned short, (__bf16)v.z);
    unsigned short u3 = __builtin_bit_cast(unsigned short, (__bf16)v.w);
    unsigned int lo = ((unsigned int)u1 << 16) | u0;
    unsigned int hi = ((unsigned int)u3 << 16) | u2;
    unsigned long long pk = ((unsigned long long)hi << 32) | lo;
    __builtin_memcpy((char*)d + i * 8, &pk, 8);
  }
}

// ---------------------------------------------------------------------------
// Fused GEMM(x2) + LSTM gate kernel.
// Block: 128 threads (4 waves). Tile: 64x64 output, all 4 gates accumulated.
// wb layout: 8 x [2048][2048] bf16 in order ii, if, ig, io, hi, hf, hg, ho.
// ---------------------------------------------------------------------------
__global__ void __launch_bounds__(128)
lstm_wmma_kernel(const __bf16* __restrict__ xb, const __bf16* __restrict__ hb,
                 const __bf16* __restrict__ wb, const float* __restrict__ cin,
                 const float* __restrict__ bias_i, const float* __restrict__ bias_f,
                 const float* __restrict__ bias_g, const float* __restrict__ bias_o,
                 float* __restrict__ out) {
  __shared__ __align__(16) __bf16 smem[2 * STAGE_ELEMS];   // 51.2 KB, double-buffered

  const int tid  = threadIdx.x;
  const int lane = tid & 31;
  const int wv   = tid >> 5;                 // wave id 0..3 -> 16-col N subtile
  const int m0   = blockIdx.x * BM;
  const int n0   = blockIdx.y * BN;

  // Low 32 bits of the flat shared address == LDS byte offset on gfx1250.
  const uint32_t lds0 = (uint32_t)(uintptr_t)(&smem[0]);

  f32x8 acc[4][4];                           // [gate][m-strip], 128 VGPRs
  const f32x8 zero = {0.f, 0.f, 0.f, 0.f, 0.f, 0.f, 0.f, 0.f};
#pragma unroll
  for (int g = 0; g < 4; ++g)
#pragma unroll
    for (int ms = 0; ms < 4; ++ms) acc[g][ms] = zero;

  // Hoist the fused bias loads (one 16-col block per wave, 4 gates).
  const int lnn   = lane & 15;
  const int khalf = lane >> 4;
  const int col   = n0 + wv * 16 + lnn;
  const float bi = bias_i[col];
  const float bf = bias_f[col];
  const float bg = bias_g[col];
  const float bo = bias_o[col];

  // Issue all async loads for K-step s into buffer `buf`.
  // Per thread: 2 x 16B chunks of the A tile, 8 x 16B chunks of the 4 B tiles.
  auto issue = [&](int s, int buf) {
    const int phase = s >> 6;                        // 0: x-phase, 1: h-phase
    const int kk    = (s & 63) * BK;
    const __bf16* Asrc = phase ? hb : xb;
    const __bf16* Wsrc = wb + (size_t)phase * 4 * HID * HID;
    const uint32_t abase = lds0 + (uint32_t)buf * (STAGE_ELEMS * 2);
#pragma unroll
    for (int i = 0; i < 2; ++i) {
      int c = tid + i * 128;                         // 256 chunks cover 64x32
      int r = c >> 2, q = c & 3;
      async_cp16(abase + (uint32_t)(r * 80 + q * 16),
                 Asrc + (size_t)(m0 + r) * HID + kk + q * 8);
    }
    const uint32_t bbase = abase + TILE_ELEMS * 2;
#pragma unroll
    for (int g = 0; g < 4; ++g) {
      const __bf16* Wg = Wsrc + (size_t)g * HID * HID;
#pragma unroll
      for (int i = 0; i < 2; ++i) {
        int c = tid + i * 128;
        int r = c >> 2, q = c & 3;
        async_cp16(bbase + (uint32_t)(g * (TILE_ELEMS * 2) + r * 80 + q * 16),
                   Wg + (size_t)(n0 + r) * HID + kk + q * 8);
      }
    }
  };

  issue(0, 0);

  for (int s = 0; s < NSTEPS; ++s) {
    const int buf = s & 1;
    if (s + 1 < NSTEPS) {
      issue(s + 1, buf ^ 1);                               // prefetch next stage
      asm volatile("s_wait_asynccnt 10" ::: "memory");     // current stage done
    } else {
      asm volatile("s_wait_asynccnt 0" ::: "memory");
    }
    __syncthreads();                                       // all waves' LDS ready

    const __bf16* As = smem + buf * STAGE_ELEMS;
    const __bf16* Bs = As + TILE_ELEMS;

    // A fragments: one per 16-row M strip (ISA layout: lane row = lnn,
    // K-half = khalf; two aligned 16B ds_load_b128 each).
    Frag a[4];
#pragma unroll
    for (int ms = 0; ms < 4; ++ms) {
      const __bf16* ap = As + (ms * 16 + lnn) * LDSTR + khalf * 8;
      __builtin_memcpy(&a[ms].q[0], ap, 16);
      __builtin_memcpy(&a[ms].q[1], ap + 16, 16);
    }

#pragma unroll
    for (int g = 0; g < 4; ++g) {
      // B fragment for this wave's column subtile (lane col = lnn,
      // K = khalf*8..+7 and 16+khalf*8..+7, matching A).
      Frag b;
      const __bf16* bp =
          Bs + g * TILE_ELEMS + (wv * 16 + lnn) * LDSTR + khalf * 8;
      __builtin_memcpy(&b.q[0], bp, 16);
      __builtin_memcpy(&b.q[1], bp + 16, 16);
#pragma unroll
      for (int ms = 0; ms < 4; ++ms) {
        acc[g][ms] = __builtin_amdgcn_wmma_f32_16x16x32_bf16(
            false, a[ms].v, false, b.v, (short)0, acc[g][ms], false, false);
      }
    }
    __syncthreads();                   // everyone done reading buf before reuse
  }

  // Fused LSTM epilogue. C/D layout: VGPR r holds (M = r + 8*khalf, N = lnn).
  const int mh = khalf * 8;
#pragma unroll
  for (int ms = 0; ms < 4; ++ms) {
#pragma unroll
    for (int r = 0; r < 8; ++r) {
      const int row = m0 + ms * 16 + mh + r;
      const float ig = fast_sigmoid(acc[0][ms][r] + bi);
      const float fg = fast_sigmoid(acc[1][ms][r] + bf);
      const float gg = fast_tanh(acc[2][ms][r] + bg);
      const float og = fast_sigmoid(acc[3][ms][r] + bo);
      const size_t idx = (size_t)row * HID + col;
      const float cn = fg * cin[idx] + ig * gg;
      const float hn = og * fast_tanh(cn);
      out[idx] = hn;                                  // h_new
      out[(size_t)BSZ * HID + idx] = cn;              // c_new
    }
  }
}

// ---------------------------------------------------------------------------
extern "C" void kernel_launch(void* const* d_in, const int* in_sizes, int n_in,
                              void* d_out, int out_size, void* d_ws, size_t ws_size,
                              hipStream_t stream) {
  (void)in_sizes; (void)n_in; (void)out_size; (void)ws_size;

  const float* x = (const float*)d_in[0];
  const float* h = (const float*)d_in[1];
  const float* c = (const float*)d_in[2];
  const float* W[8];
  for (int i = 0; i < 8; ++i) W[i] = (const float*)d_in[3 + i];  // ii,if,ig,io,hi,hf,hg,ho
  const float* bi  = (const float*)d_in[11];
  const float* bfp = (const float*)d_in[12];
  const float* bg  = (const float*)d_in[13];
  const float* bo  = (const float*)d_in[14];
  float* out = (float*)d_out;

  // Workspace layout (bf16): xb[4096*2048] | hb[4096*2048] | wb[8*2048*2048]
  __bf16* xb = (__bf16*)d_ws;
  __bf16* hb = xb + (size_t)BSZ * HID;
  __bf16* wb = hb + (size_t)BSZ * HID;

  const int nxh = BSZ * HID;   // 8M elems
  const int nw  = HID * HID;   // 4M elems
  cvt_f32_bf16<<<nxh / 1024, 256, 0, stream>>>(x, xb, nxh);
  cvt_f32_bf16<<<nxh / 1024, 256, 0, stream>>>(h, hb, nxh);
  for (int i = 0; i < 8; ++i)
    cvt_f32_bf16<<<nw / 1024, 256, 0, stream>>>(W[i], wb + (size_t)i * nw, nw);

  dim3 grid(BSZ / BM, HID / BN);   // 64 x 32 = 2048 workgroups
  lstm_wmma_kernel<<<grid, 128, 0, stream>>>(xb, hb, wb, c, bi, bfp, bg, bo, out);
}
